// FCDS_BERT_81252191306206
// MI455X (gfx1250) — matrix operand order repaired and checked by reading, not compiled, not run on hardware
//
#include <hip/hip_runtime.h>

// ---------------- problem constants ----------------
constexpr int cB = 4, cS = 512, cN = 256, cE = 42, cP = 256, cKpath = 3, cL = 4;
constexpr int cHB = 768, cDT = 20, cDI = 20, cG = 256, cNL = 2, cR = 97, cNH = 4;
constexpr int cD0   = cHB + cDT + cDI;        // 808
constexpr int cBANK = cD0 + cG * (cNL + 1);   // 1576
constexpr int cGATE = 4 * cBANK;              // 6304
constexpr int cBANKp = 1600;                  // pad of BANK
constexpr int cGATEp = 6336;                  // pad64 of GATE
constexpr int cD0p   = 832;                   // pad64 of D0
constexpr int cOF    = 5 * cBANK;             // 7880
constexpr int cOFp   = 7904;                  // pad32 of OF (K use)
constexpr int cOFn   = 7936;                  // pad64 of OF (N use)
constexpr int cM3 = cB * cP * cKpath;         // 3072
constexpr int cM1 = cB * cP;                  // 1024
constexpr int cDH = cBANK / cNH;              // 394
constexpr int cNmE = cN - cE;                 // 214

constexpr int FLAG_ACC  = 1;
constexpr int FLAG_RELU = 2;

// ---------------- types / helpers ----------------
typedef __attribute__((ext_vector_type(16))) __bf16 v16bf;
typedef __attribute__((ext_vector_type(8)))  float  v8f;
typedef __attribute__((ext_vector_type(4)))  unsigned int v4u;
typedef __attribute__((ext_vector_type(8)))  int v8i;
typedef __attribute__((ext_vector_type(4)))  int v4i;
union ABits { v16bf v; v4u q[2]; };

__device__ __forceinline__ unsigned short f2bf(float f) {
  union { float f; unsigned u; } x; x.f = f;
  unsigned u = x.u;
  u += 0x7fffu + ((u >> 16) & 1u);      // round-to-nearest-even
  return (unsigned short)(u >> 16);
}
__device__ __forceinline__ float bf2f(unsigned short h) {
  union { unsigned u; float f; } x; x.u = ((unsigned)h) << 16;
  return x.f;
}
__device__ __forceinline__ float sigm(float x) { return 1.f / (1.f + expf(-x)); }

// ---------------- TDM availability ----------------
#if defined(__gfx1250__) && __has_builtin(__builtin_amdgcn_tensor_load_to_lds) && \
    __has_builtin(__builtin_amdgcn_s_wait_tensorcnt)
#define USE_TDM 1
#else
#define USE_TDM 0
#endif

#if USE_TDM
// Issue a TDM load of a 64-row x 64-byte tile of W (row stride = Kp bf16 =
// Kp/2 dwords) into LDS at byte offset ldsOff, padding LDS rows to 96B
// (16 dwords data + 8 dwords pad) per cdna5 D# spec.
__device__ __forceinline__ void tdm_load_b_tile(const unsigned short* gsrc,
                                                unsigned ldsOff, int kpDw)
{
  unsigned long long ga = (unsigned long long)(uintptr_t)gsrc;
  v4u g0;
  g0[0] = 1u;                                            // count=1, user mode
  g0[1] = ldsOff;                                        // lds_addr (bytes)
  g0[2] = (unsigned)ga;                                  // global_addr[31:0]
  g0[3] = (unsigned)((ga >> 32) & 0x01FFFFFFu) | (2u << 30);  // addr[56:32] | type=2
  unsigned td0 = (unsigned)kpDw;                         // tensor_dim0 (dwords)
  unsigned td1 = 0x40000000u;                            // tensor_dim1: huge
  v8i g1;
  // data_size=2 (4B) | pad_enable | pad_interval=3 (16 dw) | pad_amount=7 (8 dw)
  g1[0] = (int)((2u << 16) | (1u << 20) | (3u << 22) | (7u << 25));
  g1[1] = (int)((td0 & 0xFFFFu) << 16);                  // tensor_dim0 lo
  g1[2] = (int)((td0 >> 16) | ((td1 & 0xFFFFu) << 16));  // dim0 hi | dim1 lo
  g1[3] = (int)((td1 >> 16) | (16u << 16));              // dim1 hi | tile_dim0=16
  g1[4] = (int)64u;                                      // tile_dim1=64 rows
  g1[5] = (int)td0;                                      // tensor_dim0_stride lo
  g1[6] = 0; g1[7] = 0;
  v4i z4 = {0, 0, 0, 0};
#if __clang_major__ >= 23
  v8i z8 = {0, 0, 0, 0, 0, 0, 0, 0};
  __builtin_amdgcn_tensor_load_to_lds(g0, g1, z4, z4, z8, 0);
#else
  __builtin_amdgcn_tensor_load_to_lds(g0, g1, z4, z4, 0);
#endif
}
#endif

// =====================================================================
// Core GEMM:  C[M x N] = A[M x Kp](bf16) * W[N x Kp](bf16)^T  (+bias/acc/relu)
// 128-thread block (4 waves), block tile 128M x 64N; each wave owns
// 32M x 64N = 8 x v_wmma_f32_16x16x32_bf16 per K step.  The 64x32 bf16
// B tile is staged in LDS once per block: double-buffered TDM
// (tensor_load_to_lds + s_wait_tensorcnt) when available, cooperative
// global->LDS copy otherwise.  LDS rows padded to 96B.
// Requires: M % 128 == 0, Kp % 32 == 0, W rows padded to mult of 64.
// =====================================================================
__global__ __launch_bounds__(128)
void k_gemm_bf16(const unsigned short* __restrict__ A, long long sA,
                 const unsigned short* __restrict__ W, long long sW,
                 float* __restrict__ C, long long sC,
                 int M, int Nn, int ldC, int Kp,
                 const float* __restrict__ bias, int flags)
{
  __shared__ unsigned short bst[2][64 * 48];   // 2 x 6KB B staging buffers
  const int tid  = threadIdx.x;
  const int wave = tid >> 5;
  const int lane = tid & 31;
  const int half = lane >> 4;
  const int l15  = lane & 15;
  const int tileN = blockIdx.x * 64;
  const int tileM = blockIdx.y * 128 + wave * 32;
  const int bz = blockIdx.z;
  A += (size_t)bz * sA;  W += (size_t)bz * sW;  C += (size_t)bz * sC;

  const unsigned short* Arow0 = A + (size_t)(tileM + l15) * Kp;
  const unsigned short* Arow1 = A + (size_t)(tileM + 16 + l15) * Kp;
  const unsigned short* Wtile = W + (size_t)tileN * Kp;

  v8f acc[2][4];
  if (flags & FLAG_ACC) {
#pragma unroll
    for (int mi = 0; mi < 2; ++mi)
#pragma unroll
      for (int j = 0; j < 4; ++j)
#pragma unroll
        for (int r = 0; r < 8; ++r) {
          int row = tileM + mi * 16 + r + 8 * half;
          int col = tileN + j * 16 + l15;
          acc[mi][j][r] = (row < M && col < Nn) ? C[(size_t)row * ldC + col] : 0.f;
        }
  } else {
#pragma unroll
    for (int mi = 0; mi < 2; ++mi)
#pragma unroll
      for (int j = 0; j < 4; ++j)
#pragma unroll
        for (int r = 0; r < 8; ++r) acc[mi][j][r] = 0.f;
  }

  const int nk = Kp >> 5;
#if USE_TDM
  if (tid < 32)
    tdm_load_b_tile(Wtile, (unsigned)(uintptr_t)&bst[0][0], Kp >> 1);
#endif

  for (int i = 0; i < nk; ++i) {
    const int k = i << 5;
#if USE_TDM
    if (tid < 32) __builtin_amdgcn_s_wait_tensorcnt(0);
    __syncthreads();                               // bst[i&1] ready for all waves
    if ((i + 1) < nk && tid < 32)                  // overlap next DMA with math
      tdm_load_b_tile(Wtile + (size_t)(k + 32),
                      (unsigned)(uintptr_t)&bst[(i + 1) & 1][0], Kp >> 1);
    const unsigned short* bbuf = bst[i & 1];
#else
    __syncthreads();
    {
      int r = tid >> 1, hk = tid & 1;              // 128 threads: 64 rows x 2 chunks
      *reinterpret_cast<v16bf*>(&bst[0][r * 48 + hk * 16]) =
          *reinterpret_cast<const v16bf*>(Wtile + (size_t)r * Kp + k + hk * 16);
    }
    __syncthreads();
    const unsigned short* bbuf = bst[0];
#endif
    ABits a0, a1;
    a0.q[0] = *reinterpret_cast<const v4u*>(Arow0 + k + half * 16);
    a0.q[1] = *reinterpret_cast<const v4u*>(Arow0 + k + half * 16 + 8);
    a1.q[0] = *reinterpret_cast<const v4u*>(Arow1 + k + half * 16);
    a1.q[1] = *reinterpret_cast<const v4u*>(Arow1 + k + half * 16 + 8);
    if (k + 32 < Kp) {
      __builtin_prefetch(Arow0 + k + 32, 0, 1);
      __builtin_prefetch(Arow1 + k + 32, 0, 1);
    }
#pragma unroll
    for (int j = 0; j < 4; ++j) {
      v16bf bj = *reinterpret_cast<const v16bf*>(&bbuf[(j * 16 + l15) * 48 + half * 16]);
      acc[0][j] = __builtin_amdgcn_wmma_f32_16x16x32_bf16(
          false, a0.v, false, bj, (short)0, acc[0][j], false, false);
      acc[1][j] = __builtin_amdgcn_wmma_f32_16x16x32_bf16(
          false, a1.v, false, bj, (short)0, acc[1][j], false, false);
    }
#if USE_TDM
    __syncthreads();                               // done reading before reuse
#endif
  }

#pragma unroll
  for (int mi = 0; mi < 2; ++mi)
#pragma unroll
    for (int j = 0; j < 4; ++j) {
      int col = tileN + j * 16 + l15;
      if (col >= Nn) continue;
      float bv = (bias && !(flags & FLAG_ACC)) ? bias[col] : 0.f;
#pragma unroll
      for (int r = 0; r < 8; ++r) {
        int row = tileM + mi * 16 + r + 8 * half;
        if (row >= M) continue;
        float v = acc[mi][j][r] + bv;
        if (flags & FLAG_RELU) v = fmaxf(v, 0.f);
        C[(size_t)row * ldC + col] = v;
      }
    }
}

// ---- f32 -> bf16 row-major convert with zero padding (batched by z) ----
__global__ void k_conv_bf16(const float* __restrict__ src, long long sSrc, int ldSrc,
                            int rows, int cols,
                            unsigned short* __restrict__ dst, long long sDst,
                            int ldDst, int rowsPad)
{
  long long total = (long long)rowsPad * ldDst;
  long long gid = (long long)blockIdx.x * blockDim.x + threadIdx.x;
  if (gid >= total) return;
  int c = (int)(gid % ldDst);
  int r = (int)(gid / ldDst);
  float v = (r < rows && c < cols)
              ? src[(size_t)blockIdx.z * sSrc + (size_t)r * ldSrc + c] : 0.f;
  dst[(size_t)blockIdx.z * sDst + gid] = f2bf(v);
}

// ---- f32 -> bf16 transpose-convert: dst[c][r] = src[r][c], zero pad ----
__global__ void k_transp_bf16(const float* __restrict__ src, long long sSrc, int ldSrc,
                              int rows, int cols,
                              unsigned short* __restrict__ dst, long long sDst,
                              int ldDst, int colsPad)
{
  long long total = (long long)colsPad * ldDst;
  long long gid = (long long)blockIdx.x * blockDim.x + threadIdx.x;
  if (gid >= total) return;
  int r = (int)(gid % ldDst);
  int c = (int)(gid / ldDst);
  float v = (c < cols && r < rows)
              ? src[(size_t)blockIdx.z * sSrc + (size_t)r * ldSrc + c] : 0.f;
  dst[(size_t)blockIdx.z * sDst + gid] = f2bf(v);
}

// ---- enc concat (encoder_outputs | type_emb | id_emb), stored TRANSPOSED bf16 ----
__global__ void k_build_encT(const float* __restrict__ enc, const int* __restrict__ et,
                             const int* __restrict__ eid, const float* __restrict__ temb,
                             const float* __restrict__ iemb, unsigned short* __restrict__ encT)
{
  int b = blockIdx.z;
  long long gid = (long long)blockIdx.x * blockDim.x + threadIdx.x;
  if (gid >= (long long)cD0p * cS) return;
  int s = (int)(gid % cS);
  int d = (int)(gid / cS);
  float v = 0.f;
  if (d < cHB)            v = enc[((size_t)b * cS + s) * cHB + d];
  else if (d < cHB + cDT) v = temb[et[b * cS + s] * cDT + (d - cHB)];
  else if (d < cD0)       v = iemb[eid[b * cS + s] * cDI + (d - cHB - cDT)];
  encT[((size_t)b * cD0p + d) * cS + s] = f2bf(v);
}

// ---- row softmax (row length 256) ----
__global__ void k_softmax256(float* __restrict__ x, float scale)
{
  __shared__ float sh[256];
  int row = blockIdx.x, t = threadIdx.x;
  float v = x[(size_t)row * 256 + t] * scale;
  sh[t] = v; __syncthreads();
  for (int s = 128; s; s >>= 1) { if (t < s) sh[t] = fmaxf(sh[t], sh[t + s]); __syncthreads(); }
  float mx = sh[0]; __syncthreads();
  float e = expf(v - mx);
  sh[t] = e; __syncthreads();
  for (int s = 128; s; s >>= 1) { if (t < s) sh[t] += sh[t + s]; __syncthreads(); }
  x[(size_t)row * 256 + t] = e / sh[0];
}

// ---- dst = a + b  (dst has ldDst, into feats slice) ----
__global__ void k_add2(const float* __restrict__ a, const float* __restrict__ b,
                       float* __restrict__ dst, int ldDst,
                       long long sA, long long sB, long long sD, int rows, int cols)
{
  long long total = (long long)rows * cols;
  long long gid = (long long)blockIdx.x * blockDim.x + threadIdx.x;
  if (gid >= total) return;
  int c = (int)(gid % cols), r = (int)(gid / cols);
  size_t z = blockIdx.z;
  dst[z * sD + (size_t)r * ldDst + c] =
      a[z * sA + (size_t)r * cols + c] + b[z * sB + (size_t)r * cols + c];
}

// ---- gather head/tail entity features ----
__global__ void k_gather_ht(const float* __restrict__ feats, const int* __restrict__ htp,
                            float* __restrict__ hf, float* __restrict__ tf)
{
  long long total = (long long)cM1 * cBANK;
  long long gid = (long long)blockIdx.x * blockDim.x + threadIdx.x;
  if (gid >= total) return;
  int d = (int)(gid % cBANK);
  int bp = (int)(gid / cBANK);
  int b = bp / cP;
  int hv = htp[bp * 2 + 0], tv = htp[bp * 2 + 1];
  int hi = (hv == 0) ? 0 : hv - 1;
  int ti = (tv == 0) ? 0 : tv - 1;
  hf[gid] = feats[((size_t)b * cN + cNmE + hi) * cBANK + d];
  tf[gid] = feats[((size_t)b * cN + cNmE + ti) * cBANK + d];
}

// ---- gather path node features -> bf16 A-layout per timestep ----
__global__ void k_gather_paths(const float* __restrict__ feats, const int* __restrict__ nodes,
                               unsigned short* __restrict__ pf)
{
  long long total = (long long)cM3 * cL * cBANKp;
  long long gid = (long long)blockIdx.x * blockDim.x + threadIdx.x;
  if (gid >= total) return;
  int d = (int)(gid % cBANKp);
  long long rl = gid / cBANKp;
  int l = (int)(rl % cL);
  int row = (int)(rl / cL);                  // (b*P+p)*K + k
  int b = row / (cP * cKpath);
  float v = 0.f;
  if (d < cBANK) {
    int node = nodes[(size_t)row * cL + l];
    v = feats[((size_t)b * cN + node) * cBANK + d];
  }
  pf[((size_t)l * cM3 + row) * cBANKp + d] = f2bf(v);
}

// ---- LSTM pointwise cell ----
__global__ void k_lstm_cell(const float* __restrict__ g, float* __restrict__ c,
                            unsigned short* __restrict__ hrec, unsigned short* __restrict__ hout)
{
  long long total = (long long)cM3 * cBANK;
  long long gid = (long long)blockIdx.x * blockDim.x + threadIdx.x;
  if (gid >= total) return;
  int j = (int)(gid % cBANK);
  int m = (int)(gid / cBANK);
  const float* gr = g + (size_t)m * cGATE;
  float ii = sigm(gr[j]);
  float ff = sigm(gr[cBANK + j]);
  float gg = tanhf(gr[2 * cBANK + j]);
  float oo = sigm(gr[3 * cBANK + j]);
  float cc = ff * c[gid] + ii * gg;
  c[gid] = cc;
  unsigned short hb = f2bf(oo * tanhf(cc));
  hrec[(size_t)m * cBANKp + j] = hb;
  hout[(size_t)m * cBANKp + j] = hb;
}

// ---- masked max-pool over time -> pe (bf16, padded) ----
__global__ void k_maxpool(const unsigned short* __restrict__ h2, const int* __restrict__ plens,
                          unsigned short* __restrict__ pe)
{
  long long total = (long long)cM3 * cBANKp;
  long long gid = (long long)blockIdx.x * blockDim.x + threadIdx.x;
  if (gid >= total) return;
  int j = (int)(gid % cBANKp);
  int row = (int)(gid / cBANKp);
  unsigned short out = 0;
  if (j < cBANK) {
    int len = plens[row];
    float m = -1e9f;
    for (int t = 0; t < len; ++t)
      m = fmaxf(m, bf2f(h2[((size_t)t * cM3 + row) * cBANKp + j]));
    out = f2bf(len > 0 ? m : 0.f);
  }
  pe[gid] = out;
}

// ---- query = h_f - t_f  (bf16 A-layout) ----
__global__ void k_build_qin(const float* __restrict__ hf, const float* __restrict__ tf,
                            unsigned short* __restrict__ qin)
{
  long long total = (long long)cM1 * cBANKp;
  long long gid = (long long)blockIdx.x * blockDim.x + threadIdx.x;
  if (gid >= total) return;
  int j = (int)(gid % cBANKp);
  int bp = (int)(gid / cBANKp);
  float v = (j < cBANK) ? hf[(size_t)bp * cBANK + j] - tf[(size_t)bp * cBANK + j] : 0.f;
  qin[gid] = f2bf(v);
}

// ---- tiny 3-key MHA attention: one wave per (h, p, b) ----
__global__ __launch_bounds__(32)
void k_mha_attn(const float* __restrict__ q, const float* __restrict__ kk,
                const float* __restrict__ vv, const int* __restrict__ plens,
                const float* __restrict__ rel, float* __restrict__ ctx)
{
  int h = blockIdx.x, p = blockIdx.y, b = blockIdx.z;
  int bp = b * cP + p;
  int lane = threadIdx.x;
  const float* qp = q + (size_t)bp * cBANK + h * cDH;
  float dot[cKpath];
  for (int kx = 0; kx < cKpath; ++kx) {
    const float* kp = kk + ((size_t)bp * cKpath + kx) * cBANK + h * cDH;
    float s = 0.f;
    for (int d = lane; d < cDH; d += 32) s += qp[d] * kp[d];
    for (int off = 16; off > 0; off >>= 1) s += __shfl_xor(s, off, 32);
    dot[kx] = s;
  }
  float scale = rsqrtf((float)cDH);
  bool rm = rel[bp] > 0.f;
  float sc[cKpath], mx = -3.4e38f;
  for (int kx = 0; kx < cKpath; ++kx) {
    bool valid = rm && (plens[bp * cKpath + kx] > 0);
    sc[kx] = valid ? dot[kx] * scale : -1e9f;
    mx = fmaxf(mx, sc[kx]);
  }
  float aw[cKpath], ssum = 0.f;
  for (int kx = 0; kx < cKpath; ++kx) { aw[kx] = expf(sc[kx] - mx); ssum += aw[kx]; }
  for (int kx = 0; kx < cKpath; ++kx) aw[kx] /= ssum;
  for (int d = lane; d < cDH; d += 32) {
    float s = 0.f;
    for (int kx = 0; kx < cKpath; ++kx)
      s += aw[kx] * vv[((size_t)bp * cKpath + kx) * cBANK + h * cDH + d];
    ctx[(size_t)bp * cBANK + h * cDH + d] = s;
  }
}

// ---- zero pinfo rows with no valid path ----
__global__ void k_mask_pinfo(float* __restrict__ pinfo, const int* __restrict__ plens,
                             const float* __restrict__ rel)
{
  long long total = (long long)cM1 * cBANK;
  long long gid = (long long)blockIdx.x * blockDim.x + threadIdx.x;
  if (gid >= total) return;
  int bp = (int)(gid / cBANK);
  const int* pl = plens + (size_t)bp * cKpath;
  bool any = (rel[bp] > 0.f) && (pl[0] > 0 || pl[1] > 0 || pl[2] > 0);
  if (!any) pinfo[gid] = 0.f;
}

// ---- of = [h_f | t_f | |h-t| | h*t | pinfo]  (bf16 A-layout) ----
__global__ void k_build_of(const float* __restrict__ hf, const float* __restrict__ tf,
                           const float* __restrict__ pinfo, unsigned short* __restrict__ of)
{
  long long total = (long long)cM1 * cOFp;
  long long gid = (long long)blockIdx.x * blockDim.x + threadIdx.x;
  if (gid >= total) return;
  int j = (int)(gid % cOFp);
  int bp = (int)(gid / cOFp);
  const float* h = hf + (size_t)bp * cBANK;
  const float* t = tf + (size_t)bp * cBANK;
  float v = 0.f;
  if (j < cBANK)          v = h[j];
  else if (j < 2 * cBANK) v = t[j - cBANK];
  else if (j < 3 * cBANK) v = fabsf(h[j - 2 * cBANK] - t[j - 2 * cBANK]);
  else if (j < 4 * cBANK) v = h[j - 3 * cBANK] * t[j - 3 * cBANK];
  else if (j < cOF)       v = pinfo[(size_t)bp * cBANK + (j - 4 * cBANK)];
  of[gid] = f2bf(v);
}

// =====================================================================
extern "C" void kernel_launch(void* const* d_in, const int* in_sizes, int n_in,
                              void* d_out, int out_size, void* d_ws, size_t ws_size,
                              hipStream_t stream)
{
  (void)in_sizes; (void)n_in; (void)out_size; (void)ws_size;
  // ---- inputs (setup_inputs order) ----
  const float* enc_out   = (const float*)d_in[0];
  const int*   etype     = (const int*)d_in[1];
  const int*   eid       = (const int*)d_in[2];
  const float* sub2words = (const float*)d_in[3];
  const float* adj       = (const float*)d_in[4];
  const int*   htp       = (const int*)d_in[5];
  const float* rel_mask  = (const float*)d_in[6];
  const int*   pnodes    = (const int*)d_in[7];
  const int*   plens     = (const int*)d_in[8];
  const float* temb      = (const float*)d_in[9];
  const float* iemb      = (const float*)d_in[10];
  const float* gcn0_W    = (const float*)d_in[11];
  const float* gcn0_b    = (const float*)d_in[12];
  const float* gcn_W     = (const float*)d_in[13];
  const float* gcn_b     = (const float*)d_in[14];
  const float* attn_Wq   = (const float*)d_in[15];
  const float* attn_Wk   = (const float*)d_in[16];
  const float* attn_Wv   = (const float*)d_in[17];
  const float* lstm_Wih  = (const float*)d_in[18];
  const float* lstm_Whh  = (const float*)d_in[19];
  const float* lstm_b    = (const float*)d_in[20];
  const float* mha_inW   = (const float*)d_in[21];
  const float* mha_inb   = (const float*)d_in[22];
  const float* mha_outW  = (const float*)d_in[23];
  const float* mha_outb  = (const float*)d_in[24];
  const float* pred_W    = (const float*)d_in[25];
  const float* pred_b    = (const float*)d_in[26];
  const float* mW        = (const float*)d_in[27];
  const float* mb        = (const float*)d_in[28];
  const float* bW        = (const float*)d_in[29];
  const float* bb        = (const float*)d_in[30];
  float* out = (float*)d_out;

  // ---- workspace bump allocator ----
  char* ws = (char*)d_ws;
  size_t off = 0;
  auto alloc = [&](size_t bytes) -> void* {
    off = (off + 255) & ~(size_t)255;
    void* p = ws + off; off += bytes; return p;
  };
  auto aBF = [&](long long n) { return (unsigned short*)alloc((size_t)n * 2); };
  auto aF  = [&](long long n) { return (float*)alloc((size_t)n * 4); };

  // bf16 weights (W-layout: N x Kpad, rows padded to mult of 64)
  unsigned short* gcn0Wt = aBF((long long)cG * cD0p);
  unsigned short* gcnWt  = aBF((long long)cNL * cG * cG);
  unsigned short* aqT    = aBF((long long)cNL * cG * cG);
  unsigned short* akT    = aBF((long long)cNL * cG * cG);
  unsigned short* avT    = aBF((long long)cNL * cG * cG);
  unsigned short* WihB   = aBF((long long)2 * cGATEp * cBANKp);
  unsigned short* WhhB   = aBF((long long)2 * cGATEp * cBANKp);
  unsigned short* WqB    = aBF((long long)cBANKp * cBANKp);
  unsigned short* WkB    = aBF((long long)cBANKp * cBANKp);
  unsigned short* WvB    = aBF((long long)cBANKp * cBANKp);
  unsigned short* WoB    = aBF((long long)cBANKp * cBANKp);
  unsigned short* predWt = aBF((long long)cOFn * cOFp);
  unsigned short* mWt    = aBF((long long)128 * cOFp);
  unsigned short* bWt    = aBF((long long)64 * cOFp);

  // activations
  unsigned short* encT  = aBF((long long)cB * cD0p * cS);
  unsigned short* s2wB  = aBF((long long)cB * cN * cS);
  unsigned short* adjB  = aBF((long long)cB * cN * cN);
  float*          feats = aF ((long long)cB * cN * cBANK);
  unsigned short* x0T   = aBF((long long)cB * cD0p * cN);
  float*          tmp0  = aF ((long long)cB * cN * cD0p);
  unsigned short* tmp0A = aBF((long long)cB * cN * cD0p);
  const long long g66 = (long long)cB * cN * cN;
  unsigned short *xA = aBF(g66), *xT = aBF(g66), *t1A = aBF(g66), *x1A = aBF(g66),
                 *qA = aBF(g66), *kA = aBF(g66), *attB = aBF(g66), *vT = aBF(g66);
  float *t1f = aF(g66), *x1f = aF(g66), *qf = aF(g66), *kf = aF(g66),
        *vf = aF(g66), *scf = aF(g66), *ctxf = aF(g66);
  float* h_f = aF((long long)cM1 * cBANK);
  float* t_f = aF((long long)cM1 * cBANK);
  unsigned short* pfB   = aBF((long long)cL * cM3 * cBANKp);
  unsigned short* out1B = aBF((long long)cL * cM3 * cBANKp);
  unsigned short* h2B   = aBF((long long)cL * cM3 * cBANKp);
  unsigned short* hB    = aBF((long long)cM3 * cBANKp);
  float*          cF    = aF ((long long)cM3 * cBANK);
  float*          gF    = aF ((long long)cM3 * cGATE);
  unsigned short* peB   = aBF((long long)cM3 * cBANKp);
  unsigned short* qinB  = aBF((long long)cM1 * cBANKp);
  float* qm    = aF((long long)cM1 * cBANK);
  float* km    = aF((long long)cM3 * cBANK);
  float* vm    = aF((long long)cM3 * cBANK);
  float* ctxm  = aF((long long)cM1 * cBANK);
  unsigned short* ctxB = aBF((long long)cM1 * cBANKp);
  float* pinfo = aF((long long)cM1 * cBANK);
  unsigned short* ofB  = aBF((long long)cM1 * cOFp);
  float* hidf  = aF((long long)cM1 * cOF);
  unsigned short* hidB = aBF((long long)cM1 * cOFp);

  // ---- launch helpers ----
  auto gemm = [&](const unsigned short* A, long long sA, const unsigned short* Wt, long long sW,
                  float* C, long long sC, int M, int Nn, int ldC, int Kp,
                  const float* bias, int flags, int nb) {
    dim3 grid((Nn + 63) / 64, M / 128, nb);
    k_gemm_bf16<<<grid, dim3(128), 0, stream>>>(A, sA, Wt, sW, C, sC, M, Nn, ldC, Kp, bias, flags);
  };
  auto conv = [&](const float* src, long long sS, int ldS, int rows, int cols,
                  unsigned short* dst, long long sD, int ldD, int rowsPad, int nb) {
    long long tot = (long long)rowsPad * ldD;
    dim3 grid((unsigned)((tot + 255) / 256), 1, nb);
    k_conv_bf16<<<grid, dim3(256), 0, stream>>>(src, sS, ldS, rows, cols, dst, sD, ldD, rowsPad);
  };
  auto transp = [&](const float* src, long long sS, int ldS, int rows, int cols,
                    unsigned short* dst, long long sD, int ldD, int colsPad, int nb) {
    long long tot = (long long)colsPad * ldD;
    dim3 grid((unsigned)((tot + 255) / 256), 1, nb);
    k_transp_bf16<<<grid, dim3(256), 0, stream>>>(src, sS, ldS, rows, cols, dst, sD, ldD, colsPad);
  };
  auto grid1d = [](long long tot) { return dim3((unsigned)((tot + 255) / 256)); };

  // ================= stage weights =================
  transp(gcn0_W, 0, cG, cD0, cG, gcn0Wt, 0, cD0p, cG, 1);
  for (int l = 0; l < cNL; ++l) {
    transp(gcn_W  + (size_t)l * cG * cG, 0, cG, cG, cG, gcnWt + (size_t)l * cG * cG, 0, cG, cG, 1);
    transp(attn_Wq+ (size_t)l * cG * cG, 0, cG, cG, cG, aqT   + (size_t)l * cG * cG, 0, cG, cG, 1);
    transp(attn_Wk+ (size_t)l * cG * cG, 0, cG, cG, cG, akT   + (size_t)l * cG * cG, 0, cG, cG, 1);
    transp(attn_Wv+ (size_t)l * cG * cG, 0, cG, cG, cG, avT   + (size_t)l * cG * cG, 0, cG, cG, 1);
  }
  for (int l = 0; l < 2; ++l) {
    conv(lstm_Wih + (size_t)l * cGATE * cBANK, 0, cBANK, cGATE, cBANK,
         WihB + (size_t)l * cGATEp * cBANKp, 0, cBANKp, cGATEp, 1);
    conv(lstm_Whh + (size_t)l * cGATE * cBANK, 0, cBANK, cGATE, cBANK,
         WhhB + (size_t)l * cGATEp * cBANKp, 0, cBANKp, cGATEp, 1);
  }
  conv(mha_inW,                               0, cBANK, cBANK, cBANK, WqB, 0, cBANKp, cBANKp, 1);
  conv(mha_inW + (size_t)cBANK * cBANK,       0, cBANK, cBANK, cBANK, WkB, 0, cBANKp, cBANKp, 1);
  conv(mha_inW + (size_t)2 * cBANK * cBANK,   0, cBANK, cBANK, cBANK, WvB, 0, cBANKp, cBANKp, 1);
  conv(mha_outW,                              0, cBANK, cBANK, cBANK, WoB, 0, cBANKp, cBANKp, 1);
  transp(pred_W, 0, cOF, cOF, cOF, predWt, 0, cOFp, cOFn, 1);
  transp(mW,     0, cR,  cOF, cR,  mWt,    0, cOFp, 128, 1);
  transp(bW,     0, 2,   cOF, 2,   bWt,    0, cOFp, 64,  1);

  // ================= GCN stack =================
  {
    dim3 g((unsigned)(((long long)cD0p * cS + 255) / 256), 1, cB);
    k_build_encT<<<g, dim3(256), 0, stream>>>(enc_out, etype, eid, temb, iemb, encT);
  }
  conv(sub2words, (long long)cN * cS, cS, cN, cS, s2wB, (long long)cN * cS, cS, cN, cB);
  conv(adj,       (long long)cN * cN, cN, cN, cN, adjB, (long long)cN * cN, cN, cN, cB);
  // x0 = sub2words @ enc  -> feats[:, :, 0:808]
  gemm(s2wB, (long long)cN * cS, encT, (long long)cD0p * cS,
       feats, (long long)cN * cBANK, cN, cD0, cBANK, cS, nullptr, 0, cB);
  transp(feats, (long long)cN * cBANK, cBANK, cN, cD0, x0T, (long long)cD0p * cN, cN, cD0p, cB);
  gemm(adjB, (long long)cN * cN, x0T, (long long)cD0p * cN,
       tmp0, (long long)cN * cD0p, cN, cD0, cD0p, cN, nullptr, 0, cB);
  conv(tmp0, (long long)cN * cD0p, cD0p, cN, cD0, tmp0A, (long long)cN * cD0p, cD0p, cN, cB);
  // x1 = (adj@x0) @ gcn0_W + b -> feats[:, :, 808:1064]
  gemm(tmp0A, (long long)cN * cD0p, gcn0Wt, 0,
       feats + cD0, (long long)cN * cBANK, cN, cG, cBANK, cD0p, gcn0_b, 0, cB);

  for (int l = 0; l < cNL; ++l) {
    int xoff = cD0 + cG * l, ooff = xoff + cG;
    conv (feats + xoff, (long long)cN * cBANK, cBANK, cN, cG, xA, g66 / cB, cG, cN, cB);
    transp(feats + xoff, (long long)cN * cBANK, cBANK, cN, cG, xT, g66 / cB, cG, cG, cB);
    gemm(adjB, g66 / cB, xT, g66 / cB, t1f, g66 / cB, cN, cG, cG, cG, nullptr, 0, cB);
    conv(t1f, g66 / cB, cG, cN, cG, t1A, g66 / cB, cG, cN, cB);
    gemm(t1A, g66 / cB, gcnWt + (size_t)l * cG * cG, 0,
         x1f, g66 / cB, cN, cG, cG, cG, gcn_b + (size_t)l * cG, FLAG_RELU, cB);
    conv(x1f, g66 / cB, cG, cN, cG, x1A, g66 / cB, cG, cN, cB);
    gemm(xA,  g66 / cB, aqT + (size_t)l * cG * cG, 0, qf, g66 / cB, cN, cG, cG, cG, nullptr, 0, cB);
    gemm(x1A, g66 / cB, akT + (size_t)l * cG * cG, 0, kf, g66 / cB, cN, cG, cG, cG, nullptr, 0, cB);
    gemm(x1A, g66 / cB, avT + (size_t)l * cG * cG, 0, vf, g66 / cB, cN, cG, cG, cG, nullptr, 0, cB);
    conv(qf, g66 / cB, cG, cN, cG, qA, g66 / cB, cG, cN, cB);
    conv(kf, g66 / cB, cG, cN, cG, kA, g66 / cB, cG, cN, cB);
    // sc = q @ k^T  (W-layout == k rows)
    gemm(qA, g66 / cB, kA, g66 / cB, scf, g66 / cB, cN, cN, cN, cG, nullptr, 0, cB);
    k_softmax256<<<dim3(cB * cN), dim3(256), 0, stream>>>(scf, 0.0625f);
    conv(scf, g66 / cB, cN, cN, cN, attB, g66 / cB, cN, cN, cB);
    transp(vf, g66 / cB, cG, cN, cG, vT, g66 / cB, cN, cG, cB);
    gemm(attB, g66 / cB, vT, g66 / cB, ctxf, g66 / cB, cN, cG, cG, cN, nullptr, FLAG_RELU, cB);
    {
      dim3 g((unsigned)(((long long)cN * cG + 255) / 256), 1, cB);
      k_add2<<<g, dim3(256), 0, stream>>>(x1f, ctxf, feats + ooff, cBANK,
                                          g66 / cB, g66 / cB, (long long)cN * cBANK, cN, cG);
    }
  }

  // ================= gathers =================
  k_gather_ht<<<grid1d((long long)cM1 * cBANK), dim3(256), 0, stream>>>(feats, htp, h_f, t_f);
  k_gather_paths<<<grid1d((long long)cM3 * cL * cBANKp), dim3(256), 0, stream>>>(feats, pnodes, pfB);

  // ================= 2-layer LSTM over paths =================
  hipMemsetAsync(out1B, 0, (size_t)cL * cM3 * cBANKp * 2, stream);
  hipMemsetAsync(h2B,   0, (size_t)cL * cM3 * cBANKp * 2, stream);
  for (int lay = 0; lay < 2; ++lay) {
    hipMemsetAsync(hB, 0, (size_t)cM3 * cBANKp * 2, stream);
    hipMemsetAsync(cF, 0, (size_t)cM3 * cBANK * 4, stream);
    const unsigned short* Wih = WihB + (size_t)lay * cGATEp * cBANKp;
    const unsigned short* Whh = WhhB + (size_t)lay * cGATEp * cBANKp;
    const float* bias = lstm_b + (size_t)lay * cGATE;
    for (int t = 0; t < cL; ++t) {
      const unsigned short* xin =
          (lay == 0) ? pfB + (size_t)t * cM3 * cBANKp : out1B + (size_t)t * cM3 * cBANKp;
      gemm(xin, 0, Wih, 0, gF, 0, cM3, cGATE, cGATE, cBANKp, bias, 0, 1);
      gemm(hB,  0, Whh, 0, gF, 0, cM3, cGATE, cGATE, cBANKp, nullptr, FLAG_ACC, 1);
      unsigned short* hout =
          ((lay == 0) ? out1B : h2B) + (size_t)t * cM3 * cBANKp;
      k_lstm_cell<<<grid1d((long long)cM3 * cBANK), dim3(256), 0, stream>>>(gF, cF, hB, hout);
    }
  }
  k_maxpool<<<grid1d((long long)cM3 * cBANKp), dim3(256), 0, stream>>>(h2B, plens, peB);

  // ================= MHA over paths =================
  k_build_qin<<<grid1d((long long)cM1 * cBANKp), dim3(256), 0, stream>>>(h_f, t_f, qinB);
  gemm(qinB, 0, WqB, 0, qm, 0, cM1, cBANK, cBANK, cBANKp, mha_inb,             0, 1);
  gemm(peB,  0, WkB, 0, km, 0, cM3, cBANK, cBANK, cBANKp, mha_inb + cBANK,     0, 1);
  gemm(peB,  0, WvB, 0, vm, 0, cM3, cBANK, cBANK, cBANKp, mha_inb + 2 * cBANK, 0, 1);
  k_mha_attn<<<dim3(cNH, cP, cB), dim3(32), 0, stream>>>(qm, km, vm, plens, rel_mask, ctxm);
  conv(ctxm, 0, cBANK, cM1, cBANK, ctxB, 0, cBANKp, cM1, 1);
  gemm(ctxB, 0, WoB, 0, pinfo, 0, cM1, cBANK, cBANK, cBANKp, mha_outb, 0, 1);
  k_mask_pinfo<<<grid1d((long long)cM1 * cBANK), dim3(256), 0, stream>>>(pinfo, plens, rel_mask);

  // ================= prediction head =================
  k_build_of<<<grid1d((long long)cM1 * cOFp), dim3(256), 0, stream>>>(h_f, t_f, pinfo, ofB);
  gemm(ofB, 0, predWt, 0, hidf, 0, cM1, cOF, cOF, cOFp, pred_b, FLAG_RELU, 1);
  conv(hidf, 0, cOF, cM1, cOF, hidB, 0, cOFp, cM1, 1);
  gemm(hidB, 0, mWt, 0, out,                     0, cM1, cR, cR, cOFp, mb, 0, 1);
  gemm(hidB, 0, bWt, 0, out + (size_t)cM1 * cR,  0, cM1, 2,  2,  cOFp, bb, 0, 1);
}
